// RCAMSA_70549132804208
// MI455X (gfx1250) — matrix operand order repaired and checked by reading, not compiled, not run on hardware
//
#include <hip/hip_runtime.h>
#include <hip/hip_bf16.h>
#include <cstdint>
#include <cstddef>

typedef __attribute__((ext_vector_type(16))) _Float16 v16h;
typedef __attribute__((ext_vector_type(8)))  _Float16 v8h;
typedef __attribute__((ext_vector_type(8)))  float    v8f;
typedef __attribute__((ext_vector_type(4)))  unsigned int v4u;
typedef __attribute__((ext_vector_type(8)))  int      v8i;
typedef __attribute__((ext_vector_type(4)))  int      v4i;

#define D_MODEL 1024
#define N_HEADS 16
#define SEQ     512
#define DK      64
#define BATCH   16

// ---------------------------------------------------------------------------
// WMMA fragment helpers (CDNA5 v_wmma_f32_16x16x32_f16, wave32)
// ---------------------------------------------------------------------------

// A matrix 16x32 f16, M = lane&15.  Per ISA table:
//   lanes 0-15 : VGPR0..3 -> K=0..7,  VGPR4..7 -> K=16..23
//   lanes16-31 : VGPR0..3 -> K=8..15, VGPR4..7 -> K=24..31
__device__ __forceinline__ v16h load_fragA(const _Float16* p, int ldk, int lane) {
    const int r   = lane & 15;
    const int khi = (lane >> 4) << 3;           // 0 or 8
    const _Float16* row = p + (size_t)r * ldk + khi;
    v8h lo = *(const v8h*)(row);                // K = khi .. khi+7
    v8h hi = *(const v8h*)(row + 16);           // K = 16+khi .. 16+khi+7
    v16h out;
#pragma unroll
    for (int i = 0; i < 8; ++i) { out[i] = lo[i]; out[i + 8] = hi[i]; }
    return out;
}

// B matrix 32x16 f16, N = lane&15.  Same K order as A: lanes 0-15 K=0..7,16..23; lanes 16-31 K=8..15,24..31.
// Source stored as Bt[N][K] row-major along K with leading dim ldk.
__device__ __forceinline__ v16h load_fragB(const _Float16* p, int ldk, int lane) {
    const int n  = lane & 15;
    const int kb = (lane >> 4) << 3;            // 0 or 8
    const _Float16* row = p + (size_t)n * ldk + kb;
    v8h lo = *(const v8h*)(row);
    v8h hi = *(const v8h*)(row + 16);
    v16h out;
#pragma unroll
    for (int i = 0; i < 8; ++i) { out[i] = lo[i]; out[i + 8] = hi[i]; }
    return out;
}

__device__ __forceinline__ v8f wmma_f16(v16h a, v16h b, v8f c) {
    return __builtin_amdgcn_wmma_f32_16x16x32_f16(
        /*neg_a=*/false, a, /*neg_b=*/false, b,
        /*c_mod=*/(short)0, c, /*reuse_a=*/false, /*reuse_b=*/false);
}

__device__ __forceinline__ v8f zero_v8f() {
    v8f z;
#pragma unroll
    for (int r = 0; r < 8; ++r) z[r] = 0.0f;
    return z;
}

// ---------------------------------------------------------------------------
// Tensor Data Mover: 2D tile load Global -> LDS with row padding.
// Tile: 32 f16 (64B) per row x `rows` rows; LDS row stride padded to 80B
// (pad_interval = 16 DWORDs, pad_amount = 4 DWORDs) to match [rows][40] f16.
// All descriptor inputs are wave-uniform (SGPR-resident D# groups).
// ---------------------------------------------------------------------------
__device__ __forceinline__ void tdm_load_tile_f16(unsigned lds_off,
                                                  const _Float16* gptr,
                                                  unsigned rows, unsigned kdim) {
    const unsigned long long ga = (unsigned long long)(uintptr_t)gptr;
    v4u g0;
    g0[0] = 1u;                                   // count=1 (valid), user mode
    g0[1] = lds_off;                              // lds_addr (bytes)
    g0[2] = (unsigned)(ga & 0xffffffffu);         // global_addr[31:0]
    g0[3] = (unsigned)((ga >> 32) & 0x01ffffffu)  // global_addr[56:32]
          | (2u << 30);                           // type = 2 ("image")
    v8i g1;
    g1[0] = (int)((1u << 16)                      // data_size = 2 bytes
                | (1u << 20)                      // pad_enable
                | (3u << 22)                      // pad_interval: 16 DWORDs (64B)
                | (3u << 25));                    // pad_amount:   4 DWORDs (16B)
    g1[1] = (int)((kdim & 0xffffu) << 16);        // tensor_dim0[15:0]
    g1[2] = (int)((kdim >> 16) | ((rows & 0xffffu) << 16));  // dim0 hi | dim1 lo
    g1[3] = (int)((rows >> 16) | (32u << 16));    // dim1 hi | tile_dim0 = 32
    g1[4] = (int)(rows & 0xffffu);                // tile_dim1 = rows, tile_dim2 = 0
    g1[5] = (int)kdim;                            // tensor_dim0_stride[31:0]
    g1[6] = 0;                                    // stride[47:32] | dim1_stride lo
    g1[7] = 0;
    const v4i gz4 = (v4i)0;
    const v8i gz8 = (v8i)0;
    __builtin_amdgcn_tensor_load_to_lds(g0, g1, gz4, gz4, gz8, 0);
}

// ---------------------------------------------------------------------------
// Elementwise f32 -> f16
// ---------------------------------------------------------------------------
__global__ void cvt_f32_to_f16(const float* __restrict__ in,
                               _Float16* __restrict__ out, int n) {
    int i = blockIdx.x * blockDim.x + threadIdx.x;
    const int stride = gridDim.x * blockDim.x;
    for (; i < n; i += stride) out[i] = (_Float16)in[i];
}

// ---------------------------------------------------------------------------
// Transpose f32 W[K][N] -> f16 Wt[N][K]  (LDS tiled)
// ---------------------------------------------------------------------------
__global__ __launch_bounds__(256)
void transpose_to_f16(const float* __restrict__ W, _Float16* __restrict__ Wt,
                      int K, int N) {
    __shared__ float tile[32][33];
    const int k0 = blockIdx.x * 32, n0 = blockIdx.y * 32;
    const int tx = threadIdx.x, ty = threadIdx.y;   // 32 x 8
    for (int i = ty; i < 32; i += 8)
        tile[i][tx] = W[(size_t)(k0 + i) * N + (n0 + tx)];
    __syncthreads();
    for (int i = ty; i < 32; i += 8)
        Wt[(size_t)(n0 + i) * K + (k0 + tx)] = (_Float16)tile[tx][i];
}

// ---------------------------------------------------------------------------
// Tiled WMMA GEMM:  C[M][N] = A[M][K] * Bt[N][K]^T + bias[N]
// Block tile 128x128, 256 threads (8 waves), wave tile 32x64 (2x4 WMMA tiles).
// Tiles staged to LDS by the Tensor Data Mover, double-buffered on TENSORcnt.
// mode 0: f16 out row-major
// mode 1: f32 out row-major
// mode 2: f16 out in Vt layout [B][H][DK][SEQ]   (m = b*SEQ + j, n = h*DK + d)
// ---------------------------------------------------------------------------
__global__ __launch_bounds__(256)
void gemm_wmma_f16(const _Float16* __restrict__ A, const _Float16* __restrict__ Bt,
                   const float* __restrict__ bias,
                   _Float16* __restrict__ outH, float* __restrict__ outF,
                   int M, int N, int K, int mode) {
    __shared__ _Float16 As[2][128][40];   // padded rows (80B): conflict-free b128
    __shared__ _Float16 Bs[2][128][40];

    const int tid  = threadIdx.x;
    const int lane = tid & 31;
    const int wave = tid >> 5;          // 0..7
    const int wm   = wave >> 1;         // 0..3  (M direction)
    const int wn   = wave & 1;          // 0..1  (N direction)
    const int m0   = blockIdx.y * 128;
    const int n0   = blockIdx.x * 128;
    const bool issuer = (wave == 0);    // TDM: one issuing wave (EXEC ignored)

    const unsigned ldsA0 = (unsigned)(uintptr_t)&As[0][0][0];
    const unsigned ldsA1 = (unsigned)(uintptr_t)&As[1][0][0];
    const unsigned ldsB0 = (unsigned)(uintptr_t)&Bs[0][0][0];
    const unsigned ldsB1 = (unsigned)(uintptr_t)&Bs[1][0][0];

    v8f acc[2][4];
#pragma unroll
    for (int i = 0; i < 2; ++i)
#pragma unroll
        for (int j = 0; j < 4; ++j) acc[i][j] = zero_v8f();

    const _Float16* Abase = A  + (size_t)m0 * K;
    const _Float16* Bbase = Bt + (size_t)n0 * K;

    // Prologue: stage 0 into buffer 0
    if (issuer) {
        tdm_load_tile_f16(ldsA0, Abase, 128u, (unsigned)K);
        tdm_load_tile_f16(ldsB0, Bbase, 128u, (unsigned)K);
    }

    for (int k0 = 0, it = 0; k0 < K; k0 += 32, ++it) {
        const int cur = it & 1;
        if (issuer) {
            if (k0 + 32 < K) {   // issue next stage into the other buffer
                tdm_load_tile_f16(cur ? ldsA0 : ldsA1, Abase + k0 + 32, 128u, (unsigned)K);
                tdm_load_tile_f16(cur ? ldsB0 : ldsB1, Bbase + k0 + 32, 128u, (unsigned)K);
                __builtin_amdgcn_s_wait_tensorcnt(2);   // current stage arrived
            } else {
                __builtin_amdgcn_s_wait_tensorcnt(0);   // last stage arrived
            }
        }
        __syncthreads();     // tile visible to all waves

        v16h af[2], bf[4];
#pragma unroll
        for (int i = 0; i < 2; ++i)
            af[i] = load_fragA(&As[cur][wm * 32 + i * 16][0], 40, lane);
#pragma unroll
        for (int j = 0; j < 4; ++j)
            bf[j] = load_fragB(&Bs[cur][wn * 64 + j * 16][0], 40, lane);
#pragma unroll
        for (int i = 0; i < 2; ++i)
#pragma unroll
            for (int j = 0; j < 4; ++j)
                acc[i][j] = wmma_f16(af[i], bf[j], acc[i][j]);

        __syncthreads();     // all waves done before TDM may overwrite this buffer
    }

    const int ncol = lane & 15;
    const int mtop = (lane >> 4) << 3;  // C/D rows: M = r + mtop
#pragma unroll
    for (int i = 0; i < 2; ++i) {
#pragma unroll
        for (int j = 0; j < 4; ++j) {
            const int mb = m0 + wm * 32 + i * 16 + mtop;       // first row (mult of 8)
            const int ng = n0 + wn * 64 + j * 16 + ncol;       // column
            const float bv = bias ? bias[ng] : 0.0f;
            if (mode == 2) {
                // contiguous along j: one 16B store per lane
                const int bidx = mb >> 9;                      // SEQ = 512
                const int jj   = mb & 511;
                const int h    = ng >> 6, d = ng & 63;
                v8h pk;
#pragma unroll
                for (int r = 0; r < 8; ++r) pk[r] = (_Float16)(acc[i][j][r] + bv);
                *(v8h*)(outH + (((size_t)bidx * N_HEADS + h) * DK + d) * SEQ + jj) = pk;
            } else if (mode == 1) {
#pragma unroll
                for (int r = 0; r < 8; ++r)
                    outF[(size_t)(mb + r) * N + ng] = acc[i][j][r] + bv;
            } else {
#pragma unroll
                for (int r = 0; r < 8; ++r)
                    outH[(size_t)(mb + r) * N + ng] = (_Float16)(acc[i][j][r] + bv);
            }
        }
    }
}

// ---------------------------------------------------------------------------
// Attention: per block = one (b,h) and 16 query rows.
// Phase 1: scores[16][512] = (Q Kt)*scale + rel_bias  (WMMA, into LDS f32)
// Phase 2: row softmax in LDS, probs -> f16
// Phase 3: O[16][64] = P @ V  (WMMA, split-K over wave pairs, LDS combine)
// ---------------------------------------------------------------------------
__global__ __launch_bounds__(256)
void attn_wmma(const _Float16* __restrict__ Qh, const _Float16* __restrict__ Kh,
               const _Float16* __restrict__ Vt, const float* __restrict__ rel_table,
               _Float16* __restrict__ OutH) {
    __shared__ float    Sc[16][512];      // 32 KB scores
    __shared__ _Float16 P16[16][512];     // 16 KB probs
    __shared__ float    Opart[2][16][64]; //  8 KB split-K partials

    const int tid  = threadIdx.x;
    const int lane = tid & 31;
    const int wave = tid >> 5;            // 0..7
    const int bh   = blockIdx.y;          // b*H + h
    const int b    = bh >> 4;
    const int h    = bh & 15;
    const int i0   = blockIdx.x * 16;     // query-row base

    const int ncol = lane & 15;
    const int mtop = (lane >> 4) << 3;
    const float scale = 0.125f;           // 1/sqrt(64)

    const _Float16* Qb = Qh + ((size_t)(b * SEQ + i0) * D_MODEL + h * DK);
    const v16h qf0 = load_fragA(Qb,      D_MODEL, lane);
    const v16h qf1 = load_fragA(Qb + 32, D_MODEL, lane);

    // ---- Phase 1: 32 key tiles of 16, strided over 8 waves -----------------
#pragma unroll
    for (int t = 0; t < 4; ++t) {
        const int j0 = (wave + 8 * t) * 16;
        const _Float16* Kb = Kh + ((size_t)(b * SEQ + j0) * D_MODEL + h * DK);
        v16h kf0 = load_fragB(Kb,      D_MODEL, lane);
        v16h kf1 = load_fragB(Kb + 32, D_MODEL, lane);
        v8f acc = zero_v8f();
        acc = wmma_f16(qf0, kf0, acc);
        acc = wmma_f16(qf1, kf1, acc);
        const int jc = j0 + ncol;
#pragma unroll
        for (int r = 0; r < 8; ++r) {
            const int il = mtop + r;                 // local row 0..15
            const int ig = i0 + il;                  // global query row
            const float bias = rel_table[(size_t)(ig + (SEQ - 1) - jc) * N_HEADS + h];
            Sc[il][jc] = acc[r] * scale + bias;
        }
    }
    __syncthreads();

    // ---- Phase 2: softmax, 16 lanes per row --------------------------------
    {
        const int row = tid >> 4;        // 0..15
        const int sub = tid & 15;
        float mx = -3.0e38f;
        for (int c = sub; c < 512; c += 16) mx = fmaxf(mx, Sc[row][c]);
#pragma unroll
        for (int off = 1; off < 16; off <<= 1) mx = fmaxf(mx, __shfl_xor(mx, off, 16));
        float sm = 0.0f;
        for (int c = sub; c < 512; c += 16) {
            const float e = __expf(Sc[row][c] - mx);
            Sc[row][c] = e;
            sm += e;
        }
#pragma unroll
        for (int off = 1; off < 16; off <<= 1) sm += __shfl_xor(sm, off, 16);
        const float inv = 1.0f / sm;
        for (int c = sub; c < 512; c += 16) P16[row][c] = (_Float16)(Sc[row][c] * inv);
    }
    __syncthreads();

    // ---- Phase 3: O = P @ V, 4 N-tiles x 2-way split-K ---------------------
    {
        const int nj = wave & 3;          // value-dim tile
        const int kh = wave >> 2;         // K half (j range)
        v8f accO = zero_v8f();
        const _Float16* Vb = Vt + ((size_t)bh * DK + nj * 16) * SEQ;
#pragma unroll
        for (int s = 0; s < 8; ++s) {
            const int ks = (kh * 8 + s) * 32;
            v16h pa = load_fragA(&P16[0][0] + ks, 512, lane);
            v16h vb = load_fragB(Vb + ks, SEQ, lane);
            accO = wmma_f16(pa, vb, accO);
        }
#pragma unroll
        for (int r = 0; r < 8; ++r)
            Opart[kh][mtop + r][nj * 16 + ncol] = accO[r];
    }
    __syncthreads();

    for (int e = tid; e < 16 * 64; e += 256) {
        const int rr = e >> 6, cc = e & 63;
        const float val = Opart[0][rr][cc] + Opart[1][rr][cc];
        OutH[(size_t)(b * SEQ + i0 + rr) * D_MODEL + h * DK + cc] = (_Float16)val;
    }
}

// ---------------------------------------------------------------------------
// Launch
// ---------------------------------------------------------------------------
extern "C" void kernel_launch(void* const* d_in, const int* in_sizes, int n_in,
                              void* d_out, int out_size, void* d_ws, size_t ws_size,
                              hipStream_t stream) {
    (void)in_sizes; (void)n_in; (void)out_size; (void)ws_size;
    const float* x   = (const float*)d_in[0];
    const float* Wq  = (const float*)d_in[2];
    const float* bq  = (const float*)d_in[3];
    const float* Wk  = (const float*)d_in[4];
    const float* bk  = (const float*)d_in[5];
    const float* Wv  = (const float*)d_in[6];
    const float* bv  = (const float*)d_in[7];
    const float* Wo  = (const float*)d_in[8];
    const float* bo  = (const float*)d_in[9];
    const float* rel = (const float*)d_in[10];

    char* ws = (char*)d_ws;
    size_t off = 0;
    auto take = [&](size_t bytes) -> char* {
        char* p = ws + off;
        off += (bytes + 255) & ~(size_t)255;
        return p;
    };
    const size_t BSD = (size_t)BATCH * SEQ * D_MODEL;   // 8.4M elems
    const size_t WEL = (size_t)D_MODEL * D_MODEL;

    _Float16* xh    = (_Float16*)take(BSD * 2);
    _Float16* WqT   = (_Float16*)take(WEL * 2);
    _Float16* WkT   = (_Float16*)take(WEL * 2);
    _Float16* WvT   = (_Float16*)take(WEL * 2);
    _Float16* WoT   = (_Float16*)take(WEL * 2);
    _Float16* Qh    = (_Float16*)take(BSD * 2);
    _Float16* Kh    = (_Float16*)take(BSD * 2);
    _Float16* Vt    = (_Float16*)take(BSD * 2);  // [B][H][DK][SEQ]
    _Float16* AttnH = (_Float16*)take(BSD * 2);

    // 1) x -> f16
    cvt_f32_to_f16<<<4096, 256, 0, stream>>>(x, xh, (int)BSD);

    // 2) weights -> f16, transposed to [N][K]
    {
        dim3 tb(32, 8), tg(D_MODEL / 32, D_MODEL / 32);
        transpose_to_f16<<<tg, tb, 0, stream>>>(Wq, WqT, D_MODEL, D_MODEL);
        transpose_to_f16<<<tg, tb, 0, stream>>>(Wk, WkT, D_MODEL, D_MODEL);
        transpose_to_f16<<<tg, tb, 0, stream>>>(Wv, WvT, D_MODEL, D_MODEL);
        transpose_to_f16<<<tg, tb, 0, stream>>>(Wo, WoT, D_MODEL, D_MODEL);
    }

    const int M = BATCH * SEQ;                  // 8192
    dim3 gg(D_MODEL / 128, M / 128);            // (8, 64)

    // 3) projections (WMMA + TDM-staged tiles)
    gemm_wmma_f16<<<gg, 256, 0, stream>>>(xh, WqT, bq, Qh, nullptr, M, D_MODEL, D_MODEL, 0);
    gemm_wmma_f16<<<gg, 256, 0, stream>>>(xh, WkT, bk, Kh, nullptr, M, D_MODEL, D_MODEL, 0);
    gemm_wmma_f16<<<gg, 256, 0, stream>>>(xh, WvT, bv, Vt, nullptr, M, D_MODEL, D_MODEL, 2);

    // 4) attention (WMMA + LDS softmax)
    attn_wmma<<<dim3(SEQ / 16, BATCH * N_HEADS), 256, 0, stream>>>(Qh, Kh, Vt, rel, AttnH);

    // 5) output projection -> f32 d_out (WMMA)
    gemm_wmma_f16<<<gg, 256, 0, stream>>>(AttnH, WoT, bo, nullptr, (float*)d_out,
                                          M, D_MODEL, D_MODEL, 1);
}
